// WorldModel_43825846288466
// MI455X (gfx1250) — matrix operand order repaired and checked by reading, not compiled, not run on hardware
//
#include <hip/hip_runtime.h>
#include <hip/hip_bf16.h>
#include <math.h>

typedef _Float16 v16h __attribute__((ext_vector_type(16)));
typedef _Float16 v8h  __attribute__((ext_vector_type(8)));
typedef float    v8f  __attribute__((ext_vector_type(8)));

#define BM 128
#define BN 64
#define BK 32

enum { A_DENSE = 0, A_CONV = 1, A_DECONV = 2 };
enum { S_F32ROW = 0, S_ROW16 = 1, S_NCHW16 = 2, S_RECON = 3 };

__device__ __forceinline__ float elu_f(float x) {
    return x > 0.f ? x : (expf(x) - 1.f);
}

// ---------------------------------------------------------------------------
// WMMA implicit-GEMM: C(MxN) = act( A(MxK) * B + bias )
// Geometry is compile-time. Conv (KK=4) gathers use pure shift/mask address
// math with compile-time intra-group offsets. Deconv uses a packed
// double-buffered LDS k-table (ciOff<<6 | ky<<3 | kx).
// B weights are pre-stored N-major (NP x KP, zero-padded; N padded to mult of
// 64, K to mult of 32): B-tile staging is one v8h global load + one
// ds_store_b128 per thread — no transpose scatter.
// Block: 256 threads (8 waves). Tile 128x64, K-step 32; wave = 32x32 (2x2 wmma).
// ---------------------------------------------------------------------------
template <typename TA, int AMODE, int SMODE, int ACT,
          int KP, int KR, int N, int NP,
          int Cin, int IH, int IW, int OH, int OW, int KK>
__global__ __launch_bounds__(256) void gemm_wmma_k(
        const void* __restrict__ Av, const _Float16* __restrict__ B,
        const float* __restrict__ bias, void* __restrict__ C, int M)
{
    constexpr int OHOW = OH * OW;
    constexpr int KK2  = KK * KK;
    constexpr int CHW  = IH * IW;
    static_assert(AMODE != A_CONV || KK == 4, "conv path assumes KK==4");
    static_assert(NP % BN == 0, "NP must be a multiple of the N tile");

    __shared__ __attribute__((aligned(64))) _Float16 As[BM][BK];
    __shared__ __attribute__((aligned(64))) _Float16 Bs[BN][BK];   // n-major
    __shared__ int tab[2][BK];   // deconv only: (ci*CHW)<<6 | ky<<3 | kx

    const int tid  = threadIdx.x;
    const int lane = tid & 31;
    const int wave = tid >> 5;
    const int wm   = wave & 3;
    const int wn   = wave >> 2;
    const int tileM = blockIdx.x * BM;
    const int tileN = blockIdx.y * BN;
    const int lrow  = lane & 15;

    // ---- hoisted per-thread row decomposition (loop invariant) ----
    const int rA = tid >> 1;            // A-tile row this thread fills
    const int kc = (tid & 1) * 16;      // first of 16 contiguous k's
    int m = tileM + rA;
    if (m >= M) m = M - 1;              // clamp for safe addressing; stores guarded
    const TA* Arow;
    int oy = 0, ox = 0, rowoff = 0;
    if constexpr (AMODE == A_DENSE) {
        Arow = (const TA*)Av + (long long)m * KR;
    } else {
        int n   = m / OHOW;
        int pos = m - n * OHOW;
        oy = pos / OW;
        ox = pos - oy * OW;
        Arow = (const TA*)Av + (long long)n * Cin * CHW;
        if constexpr (AMODE == A_CONV) rowoff = (oy * 2) * IW + ox * 2;
    }

    // ---- deconv k-decomposition table fill (32 threads, one k each) ----
    auto fill_tab = [&](int buf, int k0) {
        if constexpr (AMODE == A_DECONV) {
            if (tid < BK) {
                int k  = k0 + tid;
                int ci = k / KK2;
                int rr = k - ci * KK2;
                int ky = rr / KK;
                int kx = rr - ky * KK;
                tab[buf][tid] = ((ci * CHW) << 6) | (ky << 3) | kx;
            }
        }
    };

    v8f acc[2][2] = {};
    fill_tab(0, 0);
    if constexpr (AMODE == A_DECONV) __syncthreads();

    constexpr int nK = KP / BK;
    for (int kt = 0; kt < nK; ++kt) {
        const int k0  = kt * BK;
        const int buf = kt & 1;

        // --- A tile gather: 16 contiguous k's for one row ---
        {
            _Float16* dst = &As[rA][kc];
            if constexpr (AMODE == A_DENSE) {
                if constexpr (KP == KR && sizeof(TA) == 2) {
                    const v8h* pv = (const v8h*)((const _Float16*)Arow + k0 + kc);
                    *(v8h*)(dst)     = pv[0];
                    *(v8h*)(dst + 8) = pv[1];
                } else {
#pragma unroll
                    for (int j = 0; j < 16; ++j) {
                        int k = k0 + kc + j;
                        float v;
                        if constexpr (KP != KR) {
                            v = (k < KR) ? (float)Arow[k] : 0.f;
                        } else {
                            v = (float)Arow[k];
                        }
                        dst[j] = (_Float16)v;
                    }
                }
            } else if constexpr (AMODE == A_CONV) {
                // k0+kc is 16-aligned: ci constant over the 16 elements,
                // intra-group offset = (j>>2)*IW + (j&3): compile-time.
                int ciBase = (k0 + kc) >> 4;
                if constexpr (KP != KR) { if (ciBase >= Cin) ciBase = 0; }  // pad rows: B is 0
                const TA* p0 = Arow + rowoff + (long long)ciBase * CHW;
#pragma unroll
                for (int g = 0; g < 4; ++g)
#pragma unroll
                    for (int u = 0; u < 4; ++u)
                        dst[g * 4 + u] = (_Float16)(float)p0[g * IW + u];
            } else {  // A_DECONV
#pragma unroll
                for (int j = 0; j < 16; ++j) {
                    int tv    = tab[buf][kc + j];
                    int dy    = oy - ((tv >> 3) & 7);
                    int dx    = ox - (tv & 7);
                    int ciOff = tv >> 6;
                    int both  = dy | dx;
                    int iy = dy >> 1, ix = dx >> 1;
                    bool ok = (both >= 0) && ((both & 1) == 0) && (iy < IH) && (ix < IW);
                    float v = ok ? (float)Arow[ciOff + iy * IW + ix] : 0.f;
                    dst[j] = (_Float16)v;
                }
            }
        }
        // --- B tile: N-major NP x KP (zero-padded) -> n-major LDS ---
        {
            int nB   = tid >> 2;          // 0..63
            int kseg = (tid & 3) * 8;     // 0,8,16,24
            const v8h* src = (const v8h*)(B + (long long)(tileN + nB) * KP + k0 + kseg);
            *(v8h*)&Bs[nB][kseg] = *src;
        }
        fill_tab(buf ^ 1, k0 + BK);   // prefetch next table (other buffer)
        __syncthreads();

        // --- fragments: contiguous k = 16*(lane/16) + s for both A and B ---
        const int rbase = wm * 32, cbase = wn * 32;
        const int asel  = (lane < 16) ? 0 : 16;
        v16h afrag[2], bfrag[2];
#pragma unroll
        for (int mt = 0; mt < 2; ++mt) {
            const _Float16* ap = &As[rbase + mt * 16 + lrow][0];
            v8h lo = *(const v8h*)(ap + asel);
            v8h hi = *(const v8h*)(ap + asel + 8);
            afrag[mt] = __builtin_shufflevector(lo, hi, 0,1,2,3,4,5,6,7,8,9,10,11,12,13,14,15);
        }
#pragma unroll
        for (int nt = 0; nt < 2; ++nt) {
            const _Float16* bp = &Bs[cbase + nt * 16 + lrow][0] + ((lane < 16) ? 0 : 16);
            bfrag[nt] = *(const v16h*)bp;
        }
#pragma unroll
        for (int mt = 0; mt < 2; ++mt)
#pragma unroll
            for (int nt = 0; nt < 2; ++nt)
                acc[mt][nt] = __builtin_amdgcn_wmma_f32_16x16x32_f16(
                    false, afrag[mt], false, bfrag[nt], (short)0, acc[mt][nt], false, false);
        __syncthreads();
    }

    // --- epilogue: VGPR r -> M = r (+8 for lanes>=16), N = lane%16 ---
    const int rbase = wm * 32, cbase = wn * 32;
    const int radd  = (lane < 16) ? 0 : 8;
#pragma unroll
    for (int mt = 0; mt < 2; ++mt) {
#pragma unroll
        for (int nt = 0; nt < 2; ++nt) {
            int col = tileN + cbase + nt * 16 + lrow;
            if (col >= N) continue;
            float bv = bias ? bias[col] : 0.f;
#pragma unroll
            for (int r = 0; r < 8; ++r) {
                int mm = tileM + rbase + mt * 16 + radd + r;
                if (mm >= M) continue;
                float v = acc[mt][nt][r] + bv;
                if (ACT == 1) v = elu_f(v);
                if constexpr (SMODE == S_F32ROW) {
                    ((float*)C)[(long long)mm * N + col] = v;
                } else if constexpr (SMODE == S_ROW16) {
                    ((_Float16*)C)[(long long)mm * N + col] = (_Float16)v;
                } else if constexpr (SMODE == S_NCHW16) {
                    int n = mm / OHOW; int pos = mm - n * OHOW;
                    ((_Float16*)C)[((long long)n * N + col) * OHOW + pos] = (_Float16)v;
                } else {  // S_RECON: rows n = t*32 + b -> (b,t,c,pos)
                    int n = mm / OHOW; int pos = mm - n * OHOW;
                    int t = n / 32, b = n - t * 32;
                    ((float*)C)[(((long long)b * 50 + t) * N + col) * OHOW + pos] = v;
                }
            }
        }
    }
}

// ---------------------------------------------------------------------------
// Weight transforms: fp32 (N,K) row-major -> f16 N-major (Npad x Kpad), padded.
// This is a pure cast (no transpose): torch Linear/conv weights are (N,K).
// ---------------------------------------------------------------------------
__global__ void tcast_k(const float* __restrict__ W, _Float16* __restrict__ Bm,
                        int Kreal, int Kpad, int Nreal, int Npad) {
    long long i = (long long)blockIdx.x * blockDim.x + threadIdx.x;
    if (i >= (long long)Npad * Kpad) return;
    int n = (int)(i / Kpad), k = (int)(i % Kpad);
    Bm[i] = (k < Kreal && n < Nreal) ? (_Float16)W[(long long)n * Kreal + k] : (_Float16)0.f;
}

// deconv weight (Cin,Cout,kk,kk) -> Bt[co][(ci*kk+ky)*kk+kx]  (K mult of 32)
__global__ void dcast_k(const float* __restrict__ W, _Float16* __restrict__ Bm,
                        int Cin, int Cout, int kk, int Npad) {
    long long K = (long long)Cin * kk * kk;
    long long i = (long long)blockIdx.x * blockDim.x + threadIdx.x;
    if (i >= (long long)Npad * K) return;
    int co = (int)(i / K), k = (int)(i % K);
    _Float16 v = (_Float16)0.f;
    if (co < Cout) {
        int kk2 = kk * kk;
        int ci = k / kk2, r = k % kk2, ky = r / kk, kx = r % kk;
        v = (_Float16)W[(((long long)ci * Cout + co) * kk + ky) * kk + kx];
    }
    Bm[i] = v;
}

__global__ void zerof_k(float* p, long long n) {
    long long i = (long long)blockIdx.x * blockDim.x + threadIdx.x;
    if (i < n) p[i] = 0.f;
}

// ---------------------------------------------------------------------------
// RSSM per-timestep kernels (tiny, VALU)
// ---------------------------------------------------------------------------
__global__ void gru_step_k(const float* __restrict__ z, const float* __restrict__ hin,
                           const float* __restrict__ actions,
                           const float* __restrict__ wih, const float* __restrict__ whh,
                           const float* __restrict__ bih, const float* __restrict__ bhh,
                           float* __restrict__ hout, float* __restrict__ feats, int t) {
    int idx = blockIdx.x * blockDim.x + threadIdx.x;
    if (idx >= 32 * 200) return;
    int b = idx / 200, d = idx % 200;
    float x[38];
#pragma unroll
    for (int i = 0; i < 32; ++i) x[i] = z[b * 32 + i];
#pragma unroll
    for (int i = 0; i < 6; ++i) x[32 + i] = actions[((long long)b * 50 + t) * 6 + i];
    const float* h = hin + b * 200;
    float gi[3], gh[3];
#pragma unroll
    for (int gx = 0; gx < 3; ++gx) {
        int row = gx * 200 + d;
        const float* wi = wih + (long long)row * 38;
        float s = bih[row];
        for (int i = 0; i < 38; ++i) s += x[i] * wi[i];
        gi[gx] = s;
        const float* wh = whh + (long long)row * 200;
        float s2 = bhh[row];
        for (int i = 0; i < 200; ++i) s2 += h[i] * wh[i];
        gh[gx] = s2;
    }
    float r  = 1.f / (1.f + expf(-(gi[0] + gh[0])));
    float zg = 1.f / (1.f + expf(-(gi[1] + gh[1])));
    float nn = tanhf(gi[2] + r * gh[2]);
    float hv = (1.f - zg) * nn + zg * h[d];
    hout[b * 200 + d] = hv;
    feats[((long long)t * 32 + b) * 232 + 32 + d] = hv;
}

__global__ void post_hidden_k(const float* __restrict__ h, const float* __restrict__ embed,
                              const float* __restrict__ w1, const float* __restrict__ b1,
                              float* __restrict__ hid, int t) {
    int idx = blockIdx.x * blockDim.x + threadIdx.x;
    if (idx >= 32 * 200) return;
    int b = idx / 200, j = idx % 200;
    const float* w = w1 + (long long)j * 1224;
    float s = b1[j];
    const float* hb = h + b * 200;
    for (int i = 0; i < 200; ++i) s += hb[i] * w[i];
    const float* e = embed + ((long long)b * 50 + t) * 1024;
    for (int i = 0; i < 1024; ++i) s += e[i] * w[200 + i];
    hid[b * 200 + j] = elu_f(s);
}

__global__ void post_sample_k(const float* __restrict__ hid, const float* __restrict__ w2,
                              const float* __restrict__ b2, const float* __restrict__ noise,
                              float* __restrict__ z, float* __restrict__ feats, int t) {
    int idx = blockIdx.x * blockDim.x + threadIdx.x;
    if (idx >= 32 * 32) return;
    int b = idx / 32, s_ = idx % 32;
    const float* hb = hid + b * 200;
    float mean = b2[s_], lstd = b2[32 + s_];
    const float* wm = w2 + (long long)s_ * 200;
    const float* ws = w2 + (long long)(32 + s_) * 200;
    for (int i = 0; i < 200; ++i) { mean += hb[i] * wm[i]; lstd += hb[i] * ws[i]; }
    float sp  = (lstd > 20.f) ? lstd : log1pf(expf(lstd));
    float std = sp + 0.1f;
    float zv  = mean + std * noise[((long long)t * 32 + b) * 32 + s_];
    z[b * 32 + s_] = zv;
    feats[((long long)t * 32 + b) * 232 + s_] = zv;
}

__global__ void head_out_k(const float* __restrict__ hid, const float* __restrict__ w2,
                           const float* __restrict__ b2, float* __restrict__ out) {
    int m = blockIdx.x * blockDim.x + threadIdx.x;
    if (m >= 1600) return;
    float s = b2[0];
    const float* hm = hid + (long long)m * 200;
    for (int i = 0; i < 200; ++i) s += hm[i] * w2[i];
    int t = m / 32, b = m % 32;
    out[(long long)b * 50 + t] = s;
}

// ---------------------------------------------------------------------------
// Host orchestration
// ---------------------------------------------------------------------------
static inline dim3 gemm_grid(long long M, int N) {
    return dim3((unsigned)((M + BM - 1) / BM), (unsigned)((N + BN - 1) / BN));
}

extern "C" void kernel_launch(void* const* d_in, const int* in_sizes, int n_in,
                              void* d_out, int out_size, void* d_ws, size_t ws_size,
                              hipStream_t stream) {
    (void)in_sizes; (void)n_in; (void)out_size; (void)ws_size;

    const float* obs     = (const float*)d_in[0];
    const float* actions = (const float*)d_in[1];
    const float* noise   = (const float*)d_in[2];
    const float* enc_w[4] = {(const float*)d_in[3], (const float*)d_in[5], (const float*)d_in[7], (const float*)d_in[9]};
    const float* enc_b[4] = {(const float*)d_in[4], (const float*)d_in[6], (const float*)d_in[8], (const float*)d_in[10]};
    const float* embed_w = (const float*)d_in[11];
    const float* embed_b = (const float*)d_in[12];
    const float* gru_wih = (const float*)d_in[13];
    const float* gru_whh = (const float*)d_in[14];
    const float* gru_bih = (const float*)d_in[15];
    const float* gru_bhh = (const float*)d_in[16];
    // 17..20: prior_* -> dead in reference outputs, skipped
    const float* post_w1 = (const float*)d_in[21];
    const float* post_b1 = (const float*)d_in[22];
    const float* post_w2 = (const float*)d_in[23];
    const float* post_b2 = (const float*)d_in[24];
    const float* dec_fc_w = (const float*)d_in[25];
    const float* dec_fc_b = (const float*)d_in[26];
    const float* dec_w[4] = {(const float*)d_in[27], (const float*)d_in[29], (const float*)d_in[31], (const float*)d_in[33]};
    const float* dec_b[4] = {(const float*)d_in[28], (const float*)d_in[30], (const float*)d_in[32], (const float*)d_in[34]};
    const float* rew_w1 = (const float*)d_in[35]; const float* rew_b1 = (const float*)d_in[36];
    const float* rew_w2 = (const float*)d_in[37]; const float* rew_b2 = (const float*)d_in[38];
    const float* cont_w1 = (const float*)d_in[39]; const float* cont_b1 = (const float*)d_in[40];
    const float* cont_w2 = (const float*)d_in[41]; const float* cont_b2 = (const float*)d_in[42];

    // --- workspace layout (bump allocator, 256B aligned) ---
    char* base = (char*)d_ws;
    size_t off = 0;
    auto alloc = [&](size_t bytes) -> char* {
        char* p = base + off;
        off += (bytes + 255) & ~(size_t)255;
        return p;
    };
    char* bufA = alloc(147814400);   // max: d2 (1600*32*38*38 f16)
    char* bufB = alloc(59187200);    // max: d1 (1600*64*17*17 f16)
    float* embedbuf = (float*)alloc(1600ULL * 1024 * 4);
    float* feats    = (float*)alloc(50ULL * 32 * 232 * 4);
    float* hbuf     = (float*)alloc(2ULL * 32 * 200 * 4);
    float* zbuf     = (float*)alloc(32ULL * 32 * 4);
    float* phid     = (float*)alloc(32ULL * 200 * 4);
    float* headhid  = (float*)alloc(1600ULL * 200 * 4);

    // padded B matrices (N-major: NP x KP)
    int encKp[4] = {64, 512, 1024, 2048};
    int encKr[4] = {48, 512, 1024, 2048};
    int encN[4]  = {32, 64, 128, 256};
    int encNp[4] = {64, 64, 128, 256};
    _Float16* wBe[4];
    for (int l = 0; l < 4; ++l) wBe[l] = (_Float16*)alloc((size_t)encKp[l] * encNp[l] * 2);
    _Float16* wBembed = (_Float16*)alloc(1024ULL * 1024 * 2);
    _Float16* wBfc    = (_Float16*)alloc(256ULL * 1024 * 2);
    int decCin[4] = {256, 128, 64, 32}, decCout[4] = {128, 64, 32, 3}, decKK[4] = {5, 5, 6, 6};
    int decNp[4]  = {128, 64, 64, 64};
    _Float16* wBd[4];
    for (int l = 0; l < 4; ++l)
        wBd[l] = (_Float16*)alloc((size_t)decCin[l] * decKK[l] * decKK[l] * decNp[l] * 2);
    _Float16* wBrw1 = (_Float16*)alloc(256ULL * 256 * 2);
    _Float16* wBcw1 = (_Float16*)alloc(256ULL * 256 * 2);

    _Float16* x1 = (_Float16*)bufA;  // 1600x32x31x31
    _Float16* x2 = (_Float16*)bufB;  // 1600x64x14x14
    _Float16* x3 = (_Float16*)bufA;  // 1600x128x6x6
    _Float16* x4 = (_Float16*)bufB;  // 1600x256x2x2
    _Float16* dfc = (_Float16*)bufB; // 1600x1024 (x4 dead)
    _Float16* d0 = (_Float16*)bufA;  // 1600x128x7x7
    _Float16* d1 = (_Float16*)bufB;  // 1600x64x17x17
    _Float16* d2 = (_Float16*)bufA;  // 1600x32x38x38

    auto tlaunch = [&](const float* W, _Float16* Bm, int Kr, int Kp, int Nr, int Np) {
        long long n = (long long)Kp * Np;
        tcast_k<<<(unsigned)((n + 255) / 256), 256, 0, stream>>>(W, Bm, Kr, Kp, Nr, Np);
    };

    for (int l = 0; l < 4; ++l) tlaunch(enc_w[l], wBe[l], encKr[l], encKp[l], encN[l], encNp[l]);
    tlaunch(embed_w, wBembed, 1024, 1024, 1024, 1024);
    tlaunch(dec_fc_w, wBfc, 232, 256, 1024, 1024);
    for (int l = 0; l < 4; ++l) {
        long long n = (long long)decCin[l] * decKK[l] * decKK[l] * decNp[l];
        dcast_k<<<(unsigned)((n + 255) / 256), 256, 0, stream>>>(dec_w[l], wBd[l],
                                                                decCin[l], decCout[l], decKK[l], decNp[l]);
    }
    tlaunch(rew_w1, wBrw1, 232, 256, 200, 256);
    tlaunch(cont_w1, wBcw1, 232, 256, 200, 256);

    zerof_k<<<(2 * 32 * 200 + 255) / 256, 256, 0, stream>>>(hbuf, 2 * 32 * 200);
    zerof_k<<<(32 * 32 + 255) / 256, 256, 0, stream>>>(zbuf, 32 * 32);

    // --- encoder ---
    // <TA, AMODE, SMODE, ACT, KP, KR, N, NP, Cin, IH, IW, OH, OW, KK>
    gemm_wmma_k<float, A_CONV, S_NCHW16, 1, 64, 48, 32, 64, 3, 64, 64, 31, 31, 4>
        <<<gemm_grid(1600LL * 31 * 31, 32), 256, 0, stream>>>(obs, wBe[0], enc_b[0], x1, 1600 * 31 * 31);
    gemm_wmma_k<_Float16, A_CONV, S_NCHW16, 1, 512, 512, 64, 64, 32, 31, 31, 14, 14, 4>
        <<<gemm_grid(1600LL * 14 * 14, 64), 256, 0, stream>>>(x1, wBe[1], enc_b[1], x2, 1600 * 14 * 14);
    gemm_wmma_k<_Float16, A_CONV, S_NCHW16, 1, 1024, 1024, 128, 128, 64, 14, 14, 6, 6, 4>
        <<<gemm_grid(1600LL * 6 * 6, 128), 256, 0, stream>>>(x2, wBe[2], enc_b[2], x3, 1600 * 6 * 6);
    gemm_wmma_k<_Float16, A_CONV, S_NCHW16, 1, 2048, 2048, 256, 256, 128, 6, 6, 2, 2, 4>
        <<<gemm_grid(1600LL * 2 * 2, 256), 256, 0, stream>>>(x3, wBe[3], enc_b[3], x4, 1600 * 2 * 2);
    gemm_wmma_k<_Float16, A_DENSE, S_F32ROW, 0, 1024, 1024, 1024, 1024, 1, 1, 1, 1, 1, 1>
        <<<gemm_grid(1600, 1024), 256, 0, stream>>>(x4, wBembed, embed_b, embedbuf, 1600);

    // --- RSSM sequential rollout ---
    for (int t = 0; t < 50; ++t) {
        float* hin  = hbuf + (t & 1) * 6400;
        float* hout = hbuf + ((t + 1) & 1) * 6400;
        gru_step_k<<<(32 * 200 + 255) / 256, 256, 0, stream>>>(zbuf, hin, actions, gru_wih,
                                                               gru_whh, gru_bih, gru_bhh,
                                                               hout, feats, t);
        post_hidden_k<<<(32 * 200 + 255) / 256, 256, 0, stream>>>(hout, embedbuf, post_w1,
                                                                  post_b1, phid, t);
        post_sample_k<<<(32 * 32 + 255) / 256, 256, 0, stream>>>(phid, post_w2, post_b2,
                                                                 noise, zbuf, feats, t);
    }

    // --- decoder ---
    gemm_wmma_k<float, A_DENSE, S_ROW16, 0, 256, 232, 1024, 1024, 1, 1, 1, 1, 1, 1>
        <<<gemm_grid(1600, 1024), 256, 0, stream>>>(feats, wBfc, dec_fc_b, dfc, 1600);
    gemm_wmma_k<_Float16, A_DECONV, S_NCHW16, 1, 6400, 6400, 128, 128, 256, 2, 2, 7, 7, 5>
        <<<gemm_grid(1600LL * 7 * 7, 128), 256, 0, stream>>>(dfc, wBd[0], dec_b[0], d0, 1600 * 7 * 7);
    gemm_wmma_k<_Float16, A_DECONV, S_NCHW16, 1, 3200, 3200, 64, 64, 128, 7, 7, 17, 17, 5>
        <<<gemm_grid(1600LL * 17 * 17, 64), 256, 0, stream>>>(d0, wBd[1], dec_b[1], d1, 1600 * 17 * 17);
    gemm_wmma_k<_Float16, A_DECONV, S_NCHW16, 1, 2304, 2304, 32, 64, 64, 17, 17, 38, 38, 6>
        <<<gemm_grid(1600LL * 38 * 38, 32), 256, 0, stream>>>(d1, wBd[2], dec_b[2], d2, 1600 * 38 * 38);
    gemm_wmma_k<_Float16, A_DECONV, S_RECON, 0, 1152, 1152, 3, 64, 32, 38, 38, 80, 80, 6>
        <<<gemm_grid(1600LL * 80 * 80, 3), 256, 0, stream>>>(d2, wBd[3], dec_b[3], d_out, 1600 * 80 * 80);

    // --- reward / cont heads ---
    float* out_f = (float*)d_out;
    const long long recon_elems = 32LL * 50 * 3 * 80 * 80;   // 30,720,000

    gemm_wmma_k<float, A_DENSE, S_F32ROW, 1, 256, 232, 200, 256, 1, 1, 1, 1, 1, 1>
        <<<gemm_grid(1600, 200), 256, 0, stream>>>(feats, wBrw1, rew_b1, headhid, 1600);
    head_out_k<<<(1600 + 255) / 256, 256, 0, stream>>>(headhid, rew_w2, rew_b2, out_f + recon_elems);

    gemm_wmma_k<float, A_DENSE, S_F32ROW, 1, 256, 232, 200, 256, 1, 1, 1, 1, 1, 1>
        <<<gemm_grid(1600, 200), 256, 0, stream>>>(feats, wBcw1, cont_b1, headhid, 1600);
    head_out_k<<<(1600 + 255) / 256, 256, 0, stream>>>(headhid, cont_w2, cont_b2, out_f + recon_elems + 1600);
}
